// ADSA_31061203484966
// MI455X (gfx1250) — matrix operand order repaired and checked by reading, not compiled, not run on hardware
//
#include <hip/hip_runtime.h>
#include <cmath>

// ---------------------------------------------------------------------------
// MI455X (gfx1250) implementation of the Ann conv-attention block.
// All heavy GEMMs (3x3 / 1x1 convs via implicit im2col, and both attention
// einsums) run on v_wmma_f32_16x16x32_bf16 (wave32, 8-wave blocks).
// q/k/v convs are fused into a single M=768 GEMM so the shared im2col
// patches are staged once instead of three times. Weight tiles are staged
// into LDS with global_load_async_to_lds_b128 (ASYNCcnt / s_wait_asynccnt).
// ---------------------------------------------------------------------------

typedef __bf16 bf16;
typedef __attribute__((ext_vector_type(16))) __bf16 v16bf;
typedef __attribute__((ext_vector_type(8)))  __bf16 v8bf;
typedef __attribute__((ext_vector_type(8)))  float  v8f;

#define HWN 4096   // H*W
#define BATCH 8
#define CCH 256
#define NHEAD 4
#define HDIM 64
#define SZ_BCN (8ull * 256 * 4096)

// Async weight-tile staging (gfx1250 GLOBAL_LOAD_ASYNC_TO_LDS_*). Set to 0 to
// fall back to plain b128 load + ds_store staging.
#define USE_ASYNC_LDS 1

static __device__ __forceinline__ float gelu_tanh(float x) {
    float x3 = x * x * x;
    float u = 0.7978845608028654f * (x + 0.044715f * x3);
    return 0.5f * x * (1.0f + tanhf(u));
}

// ---------------------------------------------------------------------------
__global__ __launch_bounds__(256) void cast_f32_bf16(const float* __restrict__ in,
                                                     bf16* __restrict__ out, size_t n) {
    size_t i = (size_t)blockIdx.x * 256 + threadIdx.x;
    if (i < n) out[i] = (bf16)in[i];
}

// fold BN params: s = g*rsqrt(v+eps), t = b - m*s
__global__ __launch_bounds__(256) void bn_prep(const float* __restrict__ g,
                                               const float* __restrict__ b,
                                               const float* __restrict__ m,
                                               const float* __restrict__ v,
                                               float* __restrict__ s, float* __restrict__ t,
                                               int n) {
    int i = threadIdx.x;
    if (i < n) {
        float sc = g[i] * rsqrtf(v[i] + 1e-5f);
        s[i] = sc;
        t[i] = b[i] - m[i] * sc;
    }
}

// ---------------------------------------------------------------------------
// Implicit-GEMM conv (KS=3 SAME, or KS=1 pointwise) + fused BN epilogue.
// C[M=Cout, N=HWN] = W[M, K=Cin*KS*KS] x patches[K, N], per batch image.
// Block: 256 threads = 8 wave32, 128x128 output tile; wave = 64x32 sub-tile
// = 4x2 wmma accumulators.
// flags: 1=gelu 2=residual 4=write f32 8=write bf16 16=write bf16(1-y)
//        32=fused-QKV epilogue (outb=[3][BCN] bf16, outbi=[2][BCN] bf16 inv,
//                               outf=[BCN] f32 for v)
// ---------------------------------------------------------------------------
template <int KS>
__global__ __launch_bounds__(256) void conv_bn_wmma(
    const bf16* __restrict__ xb, const bf16* __restrict__ wgt,
    const float* __restrict__ bns, const float* __restrict__ bnt,
    const float* __restrict__ res, float* __restrict__ outf,
    bf16* __restrict__ outb, bf16* __restrict__ outbi,
    int Cin, int Cout, int flags) {
    const int b     = blockIdx.z;
    const int tileN = blockIdx.x * 128;
    const int tileM = blockIdx.y * 128;
    const int tid   = threadIdx.x;
    const int wave  = tid >> 5;
    const int lane  = tid & 31;
    const int waveM = wave >> 2;   // 0..1 -> 64 rows each
    const int waveN = wave & 3;    // 0..3 -> 32 cols each
    const int KK    = Cin * KS * KS;

    const int lr    = lane & 15;
    const int hi    = lane >> 4;
    const int klo   = hi ? 8 : 0;    // A-fragment K start (ISA 16-bit A layout)
    const int khalf = hi ? 16 : 0;   // unused; B fragment also starts at klo

    __shared__ __align__(16) bf16 Atile[128][32];  // [Cout rows][K]
    __shared__ __align__(16) bf16 Btile[128][32];  // [pixel cols][K]

    v8f zero = {};
    v8f acc[4][2];
#pragma unroll
    for (int i = 0; i < 4; ++i)
#pragma unroll
        for (int j = 0; j < 2; ++j) acc[i][j] = zero;

    const int arow = tid >> 1;
    const int acol = (tid & 1) << 4;
    const int j    = tid >> 1;            // pixel col this thread stages
    const int kofs = (tid & 1) << 4;
    const int pix  = tileN + j;
    const int py   = pix >> 6, px = pix & 63;
#if USE_ASYNC_LDS
    const unsigned ldsA = (unsigned)(uintptr_t)&Atile[arow][acol];
#endif

    for (int k0 = 0; k0 < KK; k0 += 32) {
        __syncthreads();
        // --- stage A: weight tile, rows are K-contiguous
        {
            const bf16* src = wgt + (size_t)(tileM + arow) * KK + k0 + acol;
#if USE_ASYNC_LDS
            // gfx1250 async DMA into LDS, tracked by ASYNCcnt
            unsigned long long ga = (unsigned long long)(uintptr_t)src;
            asm volatile("global_load_async_to_lds_b128 %0, %1, off"
                         :: "v"(ldsA), "v"(ga) : "memory");
            asm volatile("global_load_async_to_lds_b128 %0, %1, off offset:16"
                         :: "v"(ldsA), "v"(ga) : "memory");
#else
            *(v8bf*)&Atile[arow][acol]     = *(const v8bf*)(src);
            *(v8bf*)&Atile[arow][acol + 8] = *(const v8bf*)(src + 8);
#endif
            if (k0 + 32 < KK) __builtin_prefetch(src + 32, 0, 1);  // global_prefetch_b8
        }
        // --- stage B: im2col patches (zero-padded SAME)
        {
            union { bf16 e[16]; v8bf v[2]; } tmp;
#pragma unroll
            for (int i = 0; i < 16; ++i) {
                int kk = k0 + kofs + i;
                bf16 val = (bf16)0.0f;
                if (KS == 1) {
                    val = xb[((size_t)b * Cin + kk) * HWN + pix];
                } else {
                    int ci = kk / 9, r = kk - ci * 9;
                    int ky = r / 3, kx = r - ky * 3;
                    int yy = py + ky - 1, xx = px + kx - 1;
                    if (yy >= 0 && yy < 64 && xx >= 0 && xx < 64)
                        val = xb[((size_t)b * Cin + ci) * HWN + yy * 64 + xx];
                }
                tmp.e[i] = val;
            }
            *(v8bf*)&Btile[j][kofs]     = tmp.v[0];
            *(v8bf*)&Btile[j][kofs + 8] = tmp.v[1];
        }
#if USE_ASYNC_LDS
        asm volatile("s_wait_asynccnt 0" ::: "memory");
#endif
        __syncthreads();

        // --- fragments + 8 WMMAs per wave per K-step
        v16bf bfrag[2];
#pragma unroll
        for (int tn = 0; tn < 2; ++tn) {
            int col = waveN * 32 + tn * 16 + lr;
            v8bf* h = (v8bf*)&bfrag[tn];
            h[0] = *(const v8bf*)&Btile[col][klo];
            h[1] = *(const v8bf*)&Btile[col][klo + 16];
        }
#pragma unroll
        for (int tm = 0; tm < 4; ++tm) {
            int m = waveM * 64 + tm * 16 + lr;
            v16bf afrag;
            v8bf* h = (v8bf*)&afrag;
            h[0] = *(const v8bf*)&Atile[m][klo];
            h[1] = *(const v8bf*)&Atile[m][klo + 16];
#pragma unroll
            for (int tn = 0; tn < 2; ++tn)
                acc[tm][tn] = __builtin_amdgcn_wmma_f32_16x16x32_bf16(
                    false, afrag, false, bfrag[tn], (short)0, acc[tm][tn], false, false);
        }
    }

    // --- epilogue: BN (+gelu, +residual), scatter per C-fragment layout
    const bool do_gelu = (flags & 1), do_res = (flags & 2), qkv = (flags & 32);
#pragma unroll
    for (int tm = 0; tm < 4; ++tm) {
        int mbase = tileM + waveM * 64 + tm * 16 + (hi ? 8 : 0);
#pragma unroll
        for (int tn = 0; tn < 2; ++tn) {
            int n = tileN + waveN * 32 + tn * 16 + lr;
#pragma unroll
            for (int r = 0; r < 8; ++r) {
                int cout = mbase + r;
                float y = acc[tm][tn][r] * bns[cout] + bnt[cout];
                if (qkv) {
                    int sel = cout >> 8;  // 0=q 1=k 2=v
                    int c = cout & 255;
                    size_t idx = ((size_t)b * CCH + c) * HWN + n;
                    outb[(size_t)sel * SZ_BCN + idx] = (bf16)y;
                    if (sel < 2)
                        outbi[(size_t)sel * SZ_BCN + idx] = (bf16)(1.0f - y);
                    else
                        outf[idx] = y;
                } else {
                    size_t oidx = ((size_t)b * Cout + cout) * HWN + n;
                    if (do_gelu) y = gelu_tanh(y);
                    if (do_res) y += res[oidx];
                    if (flags & 4)  outf[oidx]  = y;
                    if (flags & 8)  outb[oidx]  = (bf16)y;
                    if (flags & 16) outbi[oidx] = (bf16)(1.0f - y);
                }
            }
        }
    }
}

// ---------------------------------------------------------------------------
// vT[b][h][n][e] = v[b][h*64+e][n]
// ---------------------------------------------------------------------------
__global__ __launch_bounds__(256) void transpose_v(const bf16* __restrict__ vb,
                                                   bf16* __restrict__ vT) {
    size_t idx = (size_t)blockIdx.x * 256 + threadIdx.x;  // over B*NH*N*HD
    int e = idx & 63;
    int n = (idx >> 6) & 4095;
    int h = (idx >> 18) & 3;
    int b = (int)(idx >> 20);
    vT[idx] = vb[((size_t)b * CCH + h * HDIM + e) * HWN + n];
}

// ---------------------------------------------------------------------------
// Channel attention, one block per (b, head, variant).
// Phase 1: W[64x64] = Q·K^T over N=4096 (WMMA, K-loop straight from global).
// Phase 2: out[64x4096] = (W*scale) · V (W staged bf16 in LDS, V from vT).
// ---------------------------------------------------------------------------
__global__ __launch_bounds__(256) void attn_wmma(
    const bf16* __restrict__ qa, const bf16* __restrict__ qi,
    const bf16* __restrict__ ka, const bf16* __restrict__ ki,
    const bf16* __restrict__ vT, float* __restrict__ out4) {
    const int bh = blockIdx.x;  // 0..31
    const int t  = blockIdx.y;  // variant: 0=aa 1=ii 2=ai 3=ia
    const int b = bh >> 2, h = bh & 3;
    const bf16* Q  = (t == 0 || t == 2) ? qa : qi;
    const bf16* Kp = (t == 0 || t == 3) ? ka : ki;
    float* out = out4 + (size_t)t * SZ_BCN;

    const int tid = threadIdx.x, wave = tid >> 5, lane = tid & 31;
    const int lr = lane & 15, hi = lane >> 4;
    const int klo = hi ? 8 : 0, khalf = hi ? 16 : 0;
    const size_t chbase = (size_t)b * CCH + h * HDIM;

    __shared__ __align__(16) bf16 Wlds[64][64];

    // ---- phase 1: each wave owns (m-row, 2 e-tiles) of the 4x4 tile grid
    const int mq = wave >> 1;
    const int e2 = wave & 1;
    v8f zero = {};
    v8f acc0 = zero, acc1 = zero;
    for (int n0 = 0; n0 < HWN; n0 += 32) {
        v16bf af;
        v8bf* ah = (v8bf*)&af;
        const bf16* qrow = Q + (chbase + mq * 16 + lr) * HWN + n0;
        ah[0] = *(const v8bf*)(qrow + klo);
        ah[1] = *(const v8bf*)(qrow + klo + 16);
#pragma unroll
        for (int eb = 0; eb < 2; ++eb) {
            int e = (e2 * 2 + eb) * 16 + lr;
            const bf16* krow = Kp + (chbase + e) * HWN + n0 + klo;
            v16bf bfr;
            v8bf* bh_ = (v8bf*)&bfr;
            bh_[0] = *(const v8bf*)(krow);
            bh_[1] = *(const v8bf*)(krow + 16);
            if (eb == 0)
                acc0 = __builtin_amdgcn_wmma_f32_16x16x32_bf16(false, af, false, bfr,
                                                               (short)0, acc0, false, false);
            else
                acc1 = __builtin_amdgcn_wmma_f32_16x16x32_bf16(false, af, false, bfr,
                                                               (short)0, acc1, false, false);
        }
    }
#pragma unroll
    for (int eb = 0; eb < 2; ++eb) {
        v8f a = eb ? acc1 : acc0;
        int e = (e2 * 2 + eb) * 16 + lr;
#pragma unroll
        for (int r = 0; r < 8; ++r) {
            int m = mq * 16 + (hi ? 8 : 0) + r;
            Wlds[m][e] = (bf16)(a[r] * 0.0625f);  // * C^-0.5 = 1/16
        }
    }
    __syncthreads();

    // ---- phase 2: waves stride over the 256 N-tiles
    for (int nt = wave; nt < 256; nt += 8) {
        v8f acc[4];
#pragma unroll
        for (int m = 0; m < 4; ++m) acc[m] = zero;
#pragma unroll
        for (int k0 = 0; k0 < 64; k0 += 32) {
            const bf16* vrow =
                vT + (((size_t)b * NHEAD + h) * HWN + nt * 16 + lr) * HDIM + k0 + klo;
            v16bf bfr;
            v8bf* bh_ = (v8bf*)&bfr;
            bh_[0] = *(const v8bf*)(vrow);
            bh_[1] = *(const v8bf*)(vrow + 16);
#pragma unroll
            for (int m = 0; m < 4; ++m) {
                v16bf af;
                v8bf* ah = (v8bf*)&af;
                ah[0] = *(const v8bf*)&Wlds[m * 16 + lr][k0 + klo];
                ah[1] = *(const v8bf*)&Wlds[m * 16 + lr][k0 + klo + 16];
                acc[m] = __builtin_amdgcn_wmma_f32_16x16x32_bf16(false, af, false, bfr,
                                                                 (short)0, acc[m], false, false);
            }
        }
#pragma unroll
        for (int m = 0; m < 4; ++m) {
            size_t obase = (chbase + m * 16 + (hi ? 8 : 0)) * HWN + nt * 16 + lr;
#pragma unroll
            for (int r = 0; r < 8; ++r) out[obase + (size_t)r * HWN] = acc[m][r];
        }
    }
}

// ---------------------------------------------------------------------------
// depthwise 3x3 (f32 in, bf16 out), SAME padding
// ---------------------------------------------------------------------------
__global__ __launch_bounds__(256) void dw3x3(const float* __restrict__ in,
                                             const float* __restrict__ w9,
                                             bf16* __restrict__ out) {
    size_t idx = (size_t)blockIdx.x * 256 + threadIdx.x;  // over B*C*HWN
    int pixi = idx & 4095;
    int c = (idx >> 12) & 255;
    int y = pixi >> 6, x = pixi & 63;
    const float* base = in + (idx & ~(size_t)4095);
    const float* wc = w9 + c * 9;
    float s = 0.0f;
#pragma unroll
    for (int ky = 0; ky < 3; ++ky) {
        int yy = y + ky - 1;
        if (yy < 0 || yy >= 64) continue;
#pragma unroll
        for (int kx = 0; kx < 3; ++kx) {
            int xx = x + kx - 1;
            if (xx < 0 || xx >= 64) continue;
            s += base[yy * 64 + xx] * wc[ky * 3 + kx];
        }
    }
    out[idx] = (bf16)s;
}

// cat[b,0:C] = p0+p1 ; cat[b,C:2C] = p2+p3  (bf16)
__global__ __launch_bounds__(256) void combine_cat(const float* __restrict__ p,
                                                   bf16* __restrict__ cat) {
    size_t idx = (size_t)blockIdx.x * 256 + threadIdx.x;  // over B*C*HWN
    int n = idx & 4095;
    int c = (idx >> 12) & 255;
    int b = (int)(idx >> 20);
    float a0 = p[idx] + p[SZ_BCN + idx];
    float a1 = p[2 * SZ_BCN + idx] + p[3 * SZ_BCN + idx];
    cat[((size_t)b * 512 + c) * HWN + n]       = (bf16)a0;
    cat[((size_t)b * 512 + 256 + c) * HWN + n] = (bf16)a1;
}

// norm BN: t2 = t1*s[c] + t[c]; also bf16 copy for the next WMMA conv
__global__ __launch_bounds__(256) void bn_apply(const float* __restrict__ in,
                                                const float* __restrict__ s,
                                                const float* __restrict__ t,
                                                float* __restrict__ outf,
                                                bf16* __restrict__ outb) {
    size_t idx = (size_t)blockIdx.x * 256 + threadIdx.x;
    int c = (idx >> 12) & 255;
    float y = in[idx] * s[c] + t[c];
    outf[idx] = y;
    outb[idx] = (bf16)y;
}

// ---------------------------------------------------------------------------
extern "C" void kernel_launch(void* const* d_in, const int* in_sizes, int n_in,
                              void* d_out, int out_size, void* d_ws, size_t ws_size,
                              hipStream_t stream) {
    (void)in_sizes; (void)n_in; (void)out_size; (void)ws_size;
    const float* x      = (const float*)d_in[0];
    const float* qkv_w  = (const float*)d_in[1];
    const float* qkv_g  = (const float*)d_in[2];
    const float* qkv_b  = (const float*)d_in[3];
    const float* qkv_m  = (const float*)d_in[4];
    const float* qkv_v  = (const float*)d_in[5];
    const float* dw_w   = (const float*)d_in[6];
    const float* pw_w   = (const float*)d_in[7];
    const float* ds_g   = (const float*)d_in[8];
    const float* ds_b   = (const float*)d_in[9];
    const float* ds_m   = (const float*)d_in[10];
    const float* ds_v   = (const float*)d_in[11];
    const float* fuse_w = (const float*)d_in[12];
    const float* fuse_g = (const float*)d_in[13];
    const float* fuse_b = (const float*)d_in[14];
    const float* fuse_m = (const float*)d_in[15];
    const float* fuse_v = (const float*)d_in[16];
    const float* mlp_w  = (const float*)d_in[17];
    const float* mlp_g  = (const float*)d_in[18];
    const float* mlp_b  = (const float*)d_in[19];
    const float* mlp_m  = (const float*)d_in[20];
    const float* mlp_v  = (const float*)d_in[21];
    const float* norm_g = (const float*)d_in[22];
    const float* norm_b = (const float*)d_in[23];
    const float* norm_m = (const float*)d_in[24];
    const float* norm_v = (const float*)d_in[25];
    float* out = (float*)d_out;

    const size_t BCN = SZ_BCN;  // 8.39M elems
    char* p = (char*)d_ws;
    auto alloc = [&](size_t bytes) -> void* {
        void* r = (void*)p;
        p += (bytes + 255) & ~(size_t)255;
        return r;
    };

    bf16* xb    = (bf16*)alloc(BCN * 2);
    bf16* wqkvb = (bf16*)alloc(3ull * 256 * 2304 * 2);  // stacked [768][2304]
    bf16* wpwb  = (bf16*)alloc(4ull * 256 * 256 * 2);
    bf16* wfusb = (bf16*)alloc(256ull * 4608 * 2);
    bf16* wmlpb = (bf16*)alloc(256ull * 2304 * 2);
    bf16* qkvb  = (bf16*)alloc(3 * BCN * 2);   // q, k, v bf16
    bf16* qkib  = (bf16*)alloc(2 * BCN * 2);   // 1-q, 1-k bf16
    float* vf   = (float*)alloc(BCN * 4);
    bf16* vT    = (bf16*)alloc(BCN * 2);
    float* bufA = (float*)alloc(4 * BCN * 4);  // attn outs, then reused for pw outs
    bf16* dwb   = (bf16*)alloc(4 * BCN * 2);
    bf16* catb  = (bf16*)alloc(2 * BCN * 2);
    float* t1   = (float*)alloc(BCN * 4);
    float* t2   = (float*)alloc(BCN * 4);
    bf16* t2b   = (bf16*)alloc(BCN * 2);
    float* sqkv = (float*)alloc(3 * 256 * 4);
    float* tqkv = (float*)alloc(3 * 256 * 4);
    float* sds  = (float*)alloc(4 * 256 * 4);
    float* tds  = (float*)alloc(4 * 256 * 4);
    float* sfu  = (float*)alloc(256 * 4);
    float* tfu  = (float*)alloc(256 * 4);
    float* sml  = (float*)alloc(256 * 4);
    float* tml  = (float*)alloc(256 * 4);
    float* sno  = (float*)alloc(256 * 4);
    float* tno  = (float*)alloc(256 * 4);

    const unsigned GB = (unsigned)(BCN / 256);  // 32768 blocks for BCN-sized maps

    // ---- casts ----
    cast_f32_bf16<<<GB, 256, 0, stream>>>(x, xb, BCN);
    cast_f32_bf16<<<(unsigned)((3ull * 256 * 2304 + 255) / 256), 256, 0, stream>>>(
        qkv_w, wqkvb, 3ull * 256 * 2304);
    cast_f32_bf16<<<(unsigned)((4ull * 256 * 256 + 255) / 256), 256, 0, stream>>>(
        pw_w, wpwb, 4ull * 256 * 256);
    cast_f32_bf16<<<(unsigned)((256ull * 4608 + 255) / 256), 256, 0, stream>>>(
        fuse_w, wfusb, 256ull * 4608);
    cast_f32_bf16<<<(unsigned)((256ull * 2304 + 255) / 256), 256, 0, stream>>>(
        mlp_w, wmlpb, 256ull * 2304);

    // ---- BN folds ----
    for (int i = 0; i < 3; ++i)
        bn_prep<<<1, 256, 0, stream>>>(qkv_g + i * 256, qkv_b + i * 256, qkv_m + i * 256,
                                       qkv_v + i * 256, sqkv + i * 256, tqkv + i * 256, 256);
    for (int i = 0; i < 4; ++i)
        bn_prep<<<1, 256, 0, stream>>>(ds_g + i * 256, ds_b + i * 256, ds_m + i * 256,
                                       ds_v + i * 256, sds + i * 256, tds + i * 256, 256);
    bn_prep<<<1, 256, 0, stream>>>(fuse_g, fuse_b, fuse_m, fuse_v, sfu, tfu, 256);
    bn_prep<<<1, 256, 0, stream>>>(mlp_g, mlp_b, mlp_m, mlp_v, sml, tml, 256);
    bn_prep<<<1, 256, 0, stream>>>(norm_g, norm_b, norm_m, norm_v, sno, tno, 256);

    // ---- fused q|k|v conv3x3 + BN: one GEMM, M=768 ----
    conv_bn_wmma<3><<<dim3(32, 6, 8), 256, 0, stream>>>(
        xb, wqkvb, sqkv, tqkv, nullptr, vf, qkvb, qkib, 256, 768, 32);

    // ---- attention ----
    transpose_v<<<GB, 256, 0, stream>>>(qkvb + 2 * BCN, vT);
    attn_wmma<<<dim3(32, 4), 256, 0, stream>>>(qkvb, qkib, qkvb + BCN, qkib + BCN, vT, bufA);

    // ---- ds-convs: depthwise (VALU) then pointwise WMMA GEMM + BN ----
    dim3 cgrid(32, 2, 8);
    for (int t = 0; t < 4; ++t)
        dw3x3<<<GB, 256, 0, stream>>>(bufA + (size_t)t * BCN, dw_w + (size_t)t * CCH * 9,
                                      dwb + (size_t)t * BCN);
    for (int t = 0; t < 4; ++t)
        conv_bn_wmma<1><<<cgrid, 256, 0, stream>>>(
            dwb + (size_t)t * BCN, wpwb + (size_t)t * 256 * 256, sds + t * 256, tds + t * 256,
            nullptr, bufA + (size_t)t * BCN, nullptr, nullptr, 256, 256, 4);

    // ---- cat = [aa+ii, ai+ia] (bf16, Cin=512) ----
    combine_cat<<<GB, 256, 0, stream>>>(bufA, catb);

    // ---- fuse: conv3x3(Cin=512) + BN + GELU + v residual -> t1 ----
    conv_bn_wmma<3><<<cgrid, 256, 0, stream>>>(catb, wfusb, sfu, tfu, vf, t1, nullptr,
                                               nullptr, 512, 256, 1 | 2 | 4);

    // ---- norm BN -> t2 (f32 + bf16) ----
    bn_apply<<<GB, 256, 0, stream>>>(t1, sno, tno, t2, t2b);

    // ---- mlp: conv3x3 + BN + GELU + t2 residual -> d_out ----
    conv_bn_wmma<3><<<cgrid, 256, 0, stream>>>(t2b, wmlpb, sml, tml, t2, out, nullptr,
                                               nullptr, 256, 256, 1 | 2 | 4);
}
